// SwinTransformerBlock_52329881535101
// MI455X (gfx1250) — compile-verified, matrix operand order checked by reading, not run on hardware
//
#include <hip/hip_runtime.h>
#include <hip/hip_bf16.h>

typedef __attribute__((ext_vector_type(16))) __bf16 v16bf;
typedef __attribute__((ext_vector_type(8)))  float  v8f;

#define BATCH   16
#define L_SEQ   4096
#define C_DIM   512
#define WSZ     16
#define NHEAD   4
#define NTOK    4
#define HDIM    128
#define SHIFT   8
#define NWIN    256
#define QSCALE  0.08838834764831845f   // 128^-0.5

#define QKV_TILES  96                  // 1536 / 16
#define PROJ_TILES 32                  // 512 / 16
#define KCHUNKS    16                  // 512 / 32
#define NQFRAG     (QKV_TILES * KCHUNKS * 32)   // 49152 fragments
#define NPFRAG     (PROJ_TILES * KCHUNKS * 32)  // 16384 fragments

// ---------------------------------------------------------------------------
// Prep kernel: convert fp32 weights to bf16 AND lay them out in WMMA B-
// fragment order: ws[((tile*16 + kc)*32 + lane)*16 + j].
// Fragment (tile, kc, lane) holds B[k][n] with n = tile*16 + (lane&15),
// k = kc*32 + (lane>>4)*8 + (j<8 ? j : 16 + (j-8)).
// Result: each wave's per-kc B load is 1 KB contiguous (coalesced b128s).
// ---------------------------------------------------------------------------
__global__ __launch_bounds__(256) void pack_weights(
    const float* __restrict__ qkv_w, const float* __restrict__ proj_w,
    __bf16* __restrict__ ws)
{
  int idx = blockIdx.x * 256 + threadIdx.x;     // 0 .. 65535
  const float* src;
  __bf16* dst;
  int frag;
  if (idx < NQFRAG) { frag = idx;          src = qkv_w;  dst = ws; }
  else              { frag = idx - NQFRAG; src = proj_w; dst = ws + (size_t)NQFRAG * 16; }
  int lane = frag & 31;
  int kc   = (frag >> 5) & 15;
  int tile = frag >> 9;
  int ncol = tile * 16 + (lane & 15);
  int k0   = kc * 32 + (lane >> 4) * 8;
  const float* s = src + (size_t)ncol * C_DIM + k0;
  __bf16* d = dst + (size_t)frag * 16;
  #pragma unroll
  for (int j = 0; j < 8; ++j) {
    d[j]     = (__bf16)s[j];
    d[8 + j] = (__bf16)s[16 + j];
  }
}

// Row-major fragment gather (used for GEMM2's A, which must stay row-major):
// elements 0..7 = K[k0..k0+7], elements 8..15 = K[k0+16..k0+23]
__device__ __forceinline__ v16bf load_frag(const __bf16* __restrict__ p, int k0) {
  v16bf r;
  #pragma unroll
  for (int j = 0; j < 8; ++j) { r[j] = p[k0 + j]; r[8 + j] = p[k0 + 16 + j]; }
  return r;
}

// One workgroup = 4 windows (64 tokens). 256 threads = 8 waves (wave32).
__global__ __launch_bounds__(256) void swin_block_kernel(
    const float* __restrict__ x,        const float* __restrict__ attn_mask,
    const float* __restrict__ norm_w,   const float* __restrict__ norm_b,
    const float* __restrict__ qkv_b,    const float* __restrict__ lanm,
    const float* __restrict__ proj_b,
    const __bf16* __restrict__ WqkvP,   const __bf16* __restrict__ WprojP,
    float* __restrict__ out)
{
  // 160 KB static LDS (CDNA5: up to 320 KB per workgroup)
  __shared__ __bf16 sBuf0[64 * C_DIM];            // feats_in, reused as xo (row-major)
  __shared__ float  sAcc [16 * C_DIM];            // running feat accumulator (fp32)
  __shared__ alignas(32) __bf16 sA[16 * C_DIM];   // feat in packed A-fragment order
  __shared__ __bf16 sQKV [16 * 3 * C_DIM];        // QKV result (q pre-scaled)

  const int tid   = threadIdx.x;
  const int lane  = tid & 31;
  const int wave  = tid >> 5;
  const int group = blockIdx.x;                   // 0..1023

  // ---------------- Phase 0: load + LayerNorm 64 rows ----------------
  for (int rr = 0; rr < 8; ++rr) {
    int row = wave * 8 + rr;              // 0..63
    int wl  = row >> 4;                   // window-in-group
    int t   = row & 15;                   // token-in-window
    int wg  = group * 4 + wl;
    int b   = wg >> 8;
    int nw  = wg & 255;
    int p   = nw * WSZ + t;
    int src = b * L_SEQ + ((p + SHIFT) & (L_SEQ - 1));   // roll(-SHIFT)
    const float* xr = x + (size_t)src * C_DIM;
    float xv[16];
    float s = 0.f;
    #pragma unroll
    for (int j = 0; j < 16; ++j) { xv[j] = xr[lane + j * 32]; s += xv[j]; }
    #pragma unroll
    for (int off = 16; off >= 1; off >>= 1) s += __shfl_xor(s, off, 32);
    float mean = s * (1.0f / 512.0f);
    float var = 0.f;
    #pragma unroll
    for (int j = 0; j < 16; ++j) { float d = xv[j] - mean; var += d * d; }
    #pragma unroll
    for (int off = 16; off >= 1; off >>= 1) var += __shfl_xor(var, off, 32);
    float rinv = rsqrtf(var * (1.0f / 512.0f) + 1e-5f);
    #pragma unroll
    for (int j = 0; j < 16; ++j) {
      int c = lane + j * 32;
      float val = (xv[j] - mean) * rinv * norm_w[c] + norm_b[c];
      sBuf0[row * C_DIM + c] = (__bf16)val;
    }
  }
  __syncthreads();

  // ---------------- Sequential head-group loop ----------------
  for (int it = 0; it < NHEAD; ++it) {
    // Phase 1: feat += feats_in[:, it]; write bf16 copy in packed A-frag order
    for (int e = tid; e < 16 * C_DIM; e += 256) {
      int row = e >> 9;                   // 0..15  (wl*4 + n)
      int c   = e & 511;
      int wl  = row >> 2, n = row & 3;
      float add = (float)sBuf0[(wl * 16 + it * 4 + n) * C_DIM + c];
      float acc = (it == 0) ? add : (sAcc[e] + add);
      sAcc[e] = acc;
      // packed A-fragment position for (row, c):
      int kc   = c >> 5;
      int r5   = c & 31;
      int half = (r5 >> 3) & 1;
      int j    = (r5 & 7) | ((r5 >> 4) << 3);
      int flane = half * 16 + row;
      sA[(kc * 32 + flane) * 16 + j] = (__bf16)acc;
    }
    __syncthreads();

    // Phase 2: QKV GEMM (16 x 512) @ (512 x 1536), bf16 WMMA,
    // A from packed LDS (conflict-free), B from packed global (coalesced).
    {
      const int half = lane >> 4;
      const int ln   = lane & 15;
      for (int tt = 0; tt < 12; ++tt) {
        int tileN = wave + tt * 8;        // 0..95
        int ncol  = tileN * 16 + ln;      // output feature 0..1535
        const __bf16* bbase = WqkvP + ((size_t)tileN * KCHUNKS * 32 + lane) * 16;
        v8f acc = {};
        #pragma unroll 4
        for (int kc = 0; kc < KCHUNKS; ++kc) {
          v16bf a  = *(const v16bf*)(sA + (kc * 32 + lane) * 16);
          v16bf bm = *(const v16bf*)(bbase + (size_t)kc * 32 * 16);
          acc = __builtin_amdgcn_wmma_f32_16x16x32_bf16(
              false, a, false, bm, (short)0, acc, false, false);
        }
        float bias = qkv_b[ncol];
        float sc   = (ncol < C_DIM) ? QSCALE : 1.0f;   // scale only q
        #pragma unroll
        for (int vv = 0; vv < 8; ++vv) {
          int row = half * 8 + vv;
          sQKV[row * (3 * C_DIM) + ncol] = (__bf16)((acc[vv] + bias) * sc);
        }
      }
    }
    __syncthreads();

    // Phase 3: attention per (window, head): 16 pairs x 16 threads
    {
      int pair = tid >> 4;                // 0..15
      int wl   = pair >> 2;               // window-in-group
      int h    = pair & 3;                // head
      int t16  = tid & 15;
      int n    = t16 >> 2;                // query token 0..3
      int m    = t16 & 3;                 // key token 0..3
      const __bf16* qr = sQKV + (wl * 4 + n) * (3 * C_DIM) + h * HDIM;
      const __bf16* kr = sQKV + (wl * 4 + m) * (3 * C_DIM) + C_DIM + h * HDIM;
      float dot = 0.f;
      #pragma unroll 8
      for (int d = 0; d < HDIM; ++d) dot += (float)qr[d] * (float)kr[d];
      int wg = group * 4 + wl;
      int nw = wg & 255;
      float a = dot + lanm[(n + 4 * m) * NHEAD + h]
                    + attn_mask[nw * 16 + n * 4 + m];
      // softmax over m within groups of 4 lanes
      float mx = a;
      mx = fmaxf(mx, __shfl_xor(mx, 1, 4));
      mx = fmaxf(mx, __shfl_xor(mx, 2, 4));
      float ex = __expf(a - mx);
      float sm = ex;
      sm += __shfl_xor(sm, 1, 4);
      sm += __shfl_xor(sm, 2, 4);
      float aw = ex / sm;
      float a0 = __shfl(aw, 0, 4);
      float a1 = __shfl(aw, 1, 4);
      float a2 = __shfl(aw, 2, 4);
      float a3 = __shfl(aw, 3, 4);
      // attn @ v : this lane covers query row n, d-chunk m*32..m*32+31
      const __bf16* vb = sQKV + 2 * C_DIM + h * HDIM;
      float*  accRow = sAcc  + (wl * 4 + n) * C_DIM + h * HDIM;
      __bf16* outRow = sBuf0 + (wl * 16 + it * 4 + n) * C_DIM + h * HDIM;
      int dbase = m * 32;
      #pragma unroll 4
      for (int dd = 0; dd < 32; ++dd) {
        int d = dbase + dd;
        float val = a0 * (float)vb[(wl * 4 + 0) * (3 * C_DIM) + d]
                  + a1 * (float)vb[(wl * 4 + 1) * (3 * C_DIM) + d]
                  + a2 * (float)vb[(wl * 4 + 2) * (3 * C_DIM) + d]
                  + a3 * (float)vb[(wl * 4 + 3) * (3 * C_DIM) + d];
        accRow[d] = val;            // feat for next iteration (fp32)
        outRow[d] = (__bf16)val;    // outs[it] (feats_in slot `it` is dead)
      }
    }
    __syncthreads();
  }

  // -------- Phase 4: projection + bias + residual + roll(+SHIFT) --------
  {
    const int half = lane >> 4;
    const int ln   = lane & 15;
    for (int tt = 0; tt < 16; ++tt) {
      int idx  = wave + tt * 8;           // 0..127
      int wl   = idx >> 5;                // window-in-group
      int nt   = idx & 31;                // N tile
      int ncol = nt * 16 + ln;
      const __bf16* bbase = WprojP + ((size_t)nt * KCHUNKS * 32 + lane) * 16;
      v8f acc = {};
      #pragma unroll 4
      for (int kc = 0; kc < KCHUNKS; ++kc) {
        int k0 = kc * 32 + half * 8;
        v16bf a  = load_frag(sBuf0 + (wl * 16 + ln) * C_DIM, k0);
        v16bf bm = *(const v16bf*)(bbase + (size_t)kc * 32 * 16);
        acc = __builtin_amdgcn_wmma_f32_16x16x32_bf16(
            false, a, false, bm, (short)0, acc, false, false);
      }
      float bias = proj_b[ncol];
      int wg = group * 4 + wl;
      int b  = wg >> 8;
      int nw = wg & 255;
      #pragma unroll
      for (int vv = 0; vv < 8; ++vv) {
        int t   = half * 8 + vv;          // token-in-window
        int p   = nw * WSZ + t;
        int dst = b * L_SEQ + ((p + SHIFT) & (L_SEQ - 1));  // roll(+SHIFT)
        size_t off = (size_t)dst * C_DIM + ncol;
        out[off] = x[off] + acc[vv] + bias;                 // residual
      }
    }
  }
}

extern "C" void kernel_launch(void* const* d_in, const int* in_sizes, int n_in,
                              void* d_out, int out_size, void* d_ws, size_t ws_size,
                              hipStream_t stream) {
  const float* x         = (const float*)d_in[0];
  const float* attn_mask = (const float*)d_in[1];
  const float* norm1_w   = (const float*)d_in[2];
  const float* norm1_b   = (const float*)d_in[3];
  const float* qkv_w     = (const float*)d_in[4];
  const float* qkv_b     = (const float*)d_in[5];
  const float* lanm      = (const float*)d_in[6];
  const float* proj_w    = (const float*)d_in[7];
  const float* proj_b    = (const float*)d_in[8];
  float* out = (float*)d_out;

  __bf16* ws      = (__bf16*)d_ws;
  __bf16* WqkvP   = ws;                          // 49152 * 16 bf16 = 1.5 MB
  __bf16* WprojP  = ws + (size_t)NQFRAG * 16;    // 16384 * 16 bf16 = 0.5 MB

  // (NQFRAG + NPFRAG) = 65536 fragment threads
  pack_weights<<<256, 256, 0, stream>>>(qkv_w, proj_w, ws);

  // 4096 windows / 4 per group = 1024 workgroups, 256 threads (8 waves)
  swin_block_kernel<<<1024, 256, 0, stream>>>(
      x, attn_mask, norm1_w, norm1_b, qkv_b, lanm, proj_b,
      WqkvP, WprojP, out);
}
